// RGAT_37778532335711
// MI455X (gfx1250) — matrix operand order repaired and checked by reading, not compiled, not run on hardware
//
#include <hip/hip_runtime.h>

#define D128 128
#define LRELU_SLOPE 0.2f
#define LN_EPS 1e-5f

typedef __attribute__((ext_vector_type(16))) __bf16 v16bf;
typedef __attribute__((ext_vector_type(8)))  float  v8f;

// ---------- helpers ----------
__device__ __forceinline__ float wave_sum32(float v) {
    #pragma unroll
    for (int off = 16; off > 0; off >>= 1) v += __shfl_xor(v, off, 32);
    return v;
}
// monotone float<->uint encoding for atomic max on signed floats
__device__ __forceinline__ unsigned f2ord(float f) {
    unsigned u = __float_as_uint(f);
    return (u & 0x80000000u) ? ~u : (u | 0x80000000u);
}
__device__ __forceinline__ float ord2f(unsigned k) {
    return (k & 0x80000000u) ? __uint_as_float(k & 0x7FFFFFFFu) : __uint_as_float(~k);
}

// ---------- 0) gather x = embedding[x_idx] ----------
__global__ void gather_kernel(const int* __restrict__ x_idx,
                              const float* __restrict__ emb,
                              float* __restrict__ xbuf, int N) {
    int i = blockIdx.x * blockDim.x + threadIdx.x;
    if (i >= N * D128) return;
    int n = i >> 7, d = i & 127;
    xbuf[i] = emb[(size_t)x_idx[n] * D128 + d];
}

// ---------- 1) u = Wr^T a_r ; c = br . a_r ----------
__global__ void u_kernel(const float* __restrict__ Wr, const float* __restrict__ br,
                         const float* __restrict__ a_r, float* __restrict__ u,
                         float* __restrict__ c) {
    int i = threadIdx.x;  // 128 threads
    float s = 0.f;
    for (int o = 0; o < D128; ++o) s += Wr[(size_t)o * D128 + i] * a_r[o];
    u[i] = s;
    if (i == 0) {
        float cc = 0.f;
        for (int o = 0; o < D128; ++o) cc += br[o] * a_r[o];
        *c = cc;
    }
}

// ---------- 2) rel_score[r] = rel_emb[r] . u + c ----------
__global__ void relscore_kernel(const float* __restrict__ rel_emb,
                                const float* __restrict__ u,
                                const float* __restrict__ c,
                                float* __restrict__ rs) {
    __shared__ float sh[D128];
    int r = blockIdx.x, t = threadIdx.x;
    sh[t] = rel_emb[(size_t)r * D128 + t] * u[t];
    __syncthreads();
    for (int s = 64; s > 0; s >>= 1) {
        if (t < s) sh[t] += sh[t + s];
        __syncthreads();
    }
    if (t == 0) rs[r] = sh[0] + *c;
}

// ---------- 3) hl = x @ W^T + b  (bf16 WMMA, fp32 accumulate) ----------
// block = 256 threads = 8 waves; block handles 16 rows, wave w handles cols [16w,16w+16)
__global__ __launch_bounds__(256) void hl_wmma_kernel(const float* __restrict__ X,
                                                      const float* __restrict__ W,
                                                      const float* __restrict__ b,
                                                      float* __restrict__ HL, int N) {
    const int lane = threadIdx.x & 31;
    const int wave = threadIdx.x >> 5;     // column tile 0..7
    const int rowbase = blockIdx.x * 16;
    const int mrow = lane & 15;
    const int half = lane >> 4;

    int arow = rowbase + mrow; if (arow >= N) arow = N - 1;
    const float* xrow = X + (size_t)arow * D128;
    const int ocol = wave * 16 + (lane & 15);
    const float* wrow = W + (size_t)ocol * D128;   // B column n=lane&15 -> output o

    v8f acc = {0.f, 0.f, 0.f, 0.f, 0.f, 0.f, 0.f, 0.f};

    #pragma unroll
    for (int kb = 0; kb < 4; ++kb) {
        const int k0 = kb * 32;
        // A (16x32 bf16): elems 0..7 -> K = k0 + half*8 + j ; elems 8..15 -> K = k0+16+half*8+(j-8)
        float4 a0 = *(const float4*)(xrow + k0 + half * 8);
        float4 a1 = *(const float4*)(xrow + k0 + half * 8 + 4);
        float4 a2 = *(const float4*)(xrow + k0 + 16 + half * 8);
        float4 a3 = *(const float4*)(xrow + k0 + 16 + half * 8 + 4);
        v16bf A;
        A[0]=(__bf16)a0.x; A[1]=(__bf16)a0.y; A[2]=(__bf16)a0.z; A[3]=(__bf16)a0.w;
        A[4]=(__bf16)a1.x; A[5]=(__bf16)a1.y; A[6]=(__bf16)a1.z; A[7]=(__bf16)a1.w;
        A[8]=(__bf16)a2.x; A[9]=(__bf16)a2.y; A[10]=(__bf16)a2.z; A[11]=(__bf16)a2.w;
        A[12]=(__bf16)a3.x; A[13]=(__bf16)a3.y; A[14]=(__bf16)a3.z; A[15]=(__bf16)a3.w;
        // B (32x16 bf16): elems 0..7 -> K = k0 + half*8 + j ; elems 8..15 -> K = k0+16+half*8+(j-8)
        float4 b0 = *(const float4*)(wrow + k0 + half * 8);
        float4 b1 = *(const float4*)(wrow + k0 + half * 8 + 4);
        float4 b2 = *(const float4*)(wrow + k0 + 16 + half * 8);
        float4 b3 = *(const float4*)(wrow + k0 + 16 + half * 8 + 4);
        v16bf B;
        B[0]=(__bf16)b0.x; B[1]=(__bf16)b0.y; B[2]=(__bf16)b0.z; B[3]=(__bf16)b0.w;
        B[4]=(__bf16)b1.x; B[5]=(__bf16)b1.y; B[6]=(__bf16)b1.z; B[7]=(__bf16)b1.w;
        B[8]=(__bf16)b2.x; B[9]=(__bf16)b2.y; B[10]=(__bf16)b2.z; B[11]=(__bf16)b2.w;
        B[12]=(__bf16)b3.x; B[13]=(__bf16)b3.y; B[14]=(__bf16)b3.z; B[15]=(__bf16)b3.w;

        acc = __builtin_amdgcn_wmma_f32_16x16x32_bf16(false, A, false, B,
                                                      (short)0, acc, false, false);
    }

    const float bias = b[ocol];
    #pragma unroll
    for (int r = 0; r < 8; ++r) {
        int row = rowbase + half * 8 + r;   // C/D layout: lanes 16-31 hold rows 8..15
        if (row < N) HL[(size_t)row * D128 + ocol] = acc[r] + bias;
    }
}

// ---------- 4) s_i, s_j : per-node attention scores (one wave per node) ----------
__global__ __launch_bounds__(256) void scores_kernel(const float* __restrict__ HL,
                                                     const float* __restrict__ attn,
                                                     float* __restrict__ s_i,
                                                     float* __restrict__ s_j, int N) {
    int wid = (int)((blockIdx.x * 256 + threadIdx.x) >> 5);
    int lane = threadIdx.x & 31;
    if (wid >= N) return;
    float4 hv = ((const float4*)(HL + (size_t)wid * D128))[lane];
    float4 ai = ((const float4*)attn)[lane];
    float4 aj = ((const float4*)(attn + D128))[lane];
    float si = hv.x * ai.x + hv.y * ai.y + hv.z * ai.z + hv.w * ai.w;
    float sj = hv.x * aj.x + hv.y * aj.y + hv.z * aj.z + hv.w * aj.w;
    si = wave_sum32(si);
    sj = wave_sum32(sj);
    if (lane == 0) { s_i[wid] = si; s_j[wid] = sj; }
}

// ---------- 5) zero agg, init segment state ----------
__global__ void init_kernel(float* __restrict__ agg, unsigned* __restrict__ m_enc,
                            float* __restrict__ denom, int ND, int N) {
    int i = blockIdx.x * blockDim.x + threadIdx.x;
    if (i < ND) agg[i] = 0.f;
    if (i < N) { m_enc[i] = 0u; denom[i] = 0.f; }
}

// ---------- 6) alpha + segment max ----------
__global__ void alpha_kernel(const int* __restrict__ src, const int* __restrict__ dst,
                             const int* __restrict__ etype,
                             const float* __restrict__ s_i, const float* __restrict__ s_j,
                             const float* __restrict__ rs,
                             float* __restrict__ ebuf, unsigned* __restrict__ m_enc, int E) {
    int e = blockIdx.x * blockDim.x + threadIdx.x;
    if (e >= E) return;
    int d = dst[e];
    float t = s_i[d] + s_j[src[e]] + rs[etype[e]];
    float a = t > 0.f ? t : LRELU_SLOPE * t;
    ebuf[e] = a;
    atomicMax(&m_enc[d], f2ord(a));
}

// ---------- 7) exp + segment sum ----------
__global__ void expsum_kernel(const int* __restrict__ dst, const unsigned* __restrict__ m_enc,
                              float* __restrict__ ebuf, float* __restrict__ denom, int E) {
    int e = blockIdx.x * blockDim.x + threadIdx.x;
    if (e >= E) return;
    int d = dst[e];
    float v = __expf(ebuf[e] - ord2f(m_enc[d]));
    ebuf[e] = v;
    atomicAdd(&denom[d], v);
}

// ---------- 8) weighted message scatter (one wave per edge) ----------
__global__ __launch_bounds__(256) void scatter_kernel(const int* __restrict__ src,
                                                      const int* __restrict__ dst,
                                                      const float* __restrict__ ebuf,
                                                      const float* __restrict__ denom,
                                                      const float* __restrict__ HL,
                                                      float* __restrict__ agg, int E) {
    int wid = (int)((blockIdx.x * 256 + threadIdx.x) >> 5);
    int lane = threadIdx.x & 31;
    if (wid >= E) return;
    int s = src[wid], d = dst[wid];
    float w = ebuf[wid] / denom[d];
    float4 hv = ((const float4*)(HL + (size_t)s * D128))[lane];
    float* out = agg + (size_t)d * D128 + lane * 4;
    atomicAdd(out + 0, hv.x * w);
    atomicAdd(out + 1, hv.y * w);
    atomicAdd(out + 2, hv.z * w);
    atomicAdd(out + 3, hv.w * w);
}

// ---------- 9) residual + layernorm + relu (one wave per node) ----------
__global__ __launch_bounds__(256) void norm_kernel(const float* __restrict__ X,
                                                   const float* __restrict__ agg,
                                                   const float* __restrict__ g,
                                                   const float* __restrict__ be,
                                                   float* __restrict__ out, int N) {
    int wid = (int)((blockIdx.x * 256 + threadIdx.x) >> 5);
    int lane = threadIdx.x & 31;
    if (wid >= N) return;
    float4 xv = ((const float4*)(X + (size_t)wid * D128))[lane];
    float4 av = ((const float4*)(agg + (size_t)wid * D128))[lane];
    float y0 = xv.x + av.x, y1 = xv.y + av.y, y2 = xv.z + av.z, y3 = xv.w + av.w;
    float mu = wave_sum32(y0 + y1 + y2 + y3) * (1.f / 128.f);
    float d0 = y0 - mu, d1 = y1 - mu, d2 = y2 - mu, d3 = y3 - mu;
    float var = wave_sum32(d0 * d0 + d1 * d1 + d2 * d2 + d3 * d3) * (1.f / 128.f);
    float inv = rsqrtf(var + LN_EPS);
    float4 gv = ((const float4*)g)[lane];
    float4 bv = ((const float4*)be)[lane];
    float4 res;
    res.x = fmaxf(d0 * inv * gv.x + bv.x, 0.f);
    res.y = fmaxf(d1 * inv * gv.y + bv.y, 0.f);
    res.z = fmaxf(d2 * inv * gv.z + bv.z, 0.f);
    res.w = fmaxf(d3 * inv * gv.w + bv.w, 0.f);
    ((float4*)(out + (size_t)wid * D128))[lane] = res;
}

// ---------- host side ----------
static void run_layer(const float* xin, float* xout,
                      const int* src, const int* dst, const int* etype,
                      const float* rel_emb,
                      const float* W, const float* b,
                      const float* Wr, const float* br,
                      const float* attn, const float* g, const float* be,
                      float* hl, float* agg, float* ebuf,
                      float* s_i, float* s_j, unsigned* m_enc, float* denom,
                      float* rs, float* u, float* c,
                      int N, int E, int R, hipStream_t stream) {
    const int ND = N * D128;
    u_kernel<<<1, 128, 0, stream>>>(Wr, br, attn + 2 * D128, u, c);
    relscore_kernel<<<R, 128, 0, stream>>>(rel_emb, u, c, rs);
    hl_wmma_kernel<<<(N + 15) / 16, 256, 0, stream>>>(xin, W, b, hl, N);
    scores_kernel<<<(N + 7) / 8, 256, 0, stream>>>(hl, attn, s_i, s_j, N);
    init_kernel<<<(ND + 255) / 256, 256, 0, stream>>>(agg, m_enc, denom, ND, N);
    alpha_kernel<<<(E + 255) / 256, 256, 0, stream>>>(src, dst, etype, s_i, s_j, rs, ebuf, m_enc, E);
    expsum_kernel<<<(E + 255) / 256, 256, 0, stream>>>(dst, m_enc, ebuf, denom, E);
    scatter_kernel<<<(E + 7) / 8, 256, 0, stream>>>(src, dst, ebuf, denom, hl, agg, E);
    norm_kernel<<<(N + 7) / 8, 256, 0, stream>>>(xin, agg, g, be, xout, N);
}

extern "C" void kernel_launch(void* const* d_in, const int* in_sizes, int n_in,
                              void* d_out, int out_size, void* d_ws, size_t ws_size,
                              hipStream_t stream) {
    const int N = in_sizes[0];
    const int E = in_sizes[2];
    const int R = in_sizes[4] / D128;
    const int ND = N * D128;

    const int*   x_idx   = (const int*)d_in[0];
    const int*   src     = (const int*)d_in[1];          // edge_index[0]
    const int*   dst     = src + E;                      // edge_index[1]
    const int*   etype   = (const int*)d_in[2];
    const float* emb     = (const float*)d_in[3];
    const float* rel_emb = (const float*)d_in[4];
    const float* W1 = (const float*)d_in[5],  *b1 = (const float*)d_in[6];
    const float* Wr1 = (const float*)d_in[7], *br1 = (const float*)d_in[8];
    const float* attn1 = (const float*)d_in[9];
    const float* g1 = (const float*)d_in[10], *be1 = (const float*)d_in[11];
    const float* W2 = (const float*)d_in[12], *b2 = (const float*)d_in[13];
    const float* Wr2 = (const float*)d_in[14], *br2 = (const float*)d_in[15];
    const float* attn2 = (const float*)d_in[16];
    const float* g2 = (const float*)d_in[17], *be2 = (const float*)d_in[18];

    // workspace carve-out
    char* p = (char*)d_ws;
    auto carve = [&](size_t bytes) -> void* {
        void* r = (void*)p;
        p += (bytes + 255) & ~(size_t)255;
        return r;
    };
    float*    xbuf  = (float*)carve((size_t)ND * 4);
    float*    hl    = (float*)carve((size_t)ND * 4);
    float*    agg   = (float*)carve((size_t)ND * 4);
    float*    ebuf  = (float*)carve((size_t)E * 4);
    float*    s_i   = (float*)carve((size_t)N * 4);
    float*    s_j   = (float*)carve((size_t)N * 4);
    unsigned* m_enc = (unsigned*)carve((size_t)N * 4);
    float*    denom = (float*)carve((size_t)N * 4);
    float*    rs    = (float*)carve((size_t)R * 4);
    float*    u     = (float*)carve((size_t)D128 * 4);
    float*    c     = (float*)carve(256);

    gather_kernel<<<(ND + 255) / 256, 256, 0, stream>>>(x_idx, emb, xbuf, N);

    // layer 1: write layernorm result back into xbuf (per-thread read-before-write, safe)
    run_layer(xbuf, xbuf, src, dst, etype, rel_emb, W1, b1, Wr1, br1, attn1, g1, be1,
              hl, agg, ebuf, s_i, s_j, m_enc, denom, rs, u, c, N, E, R, stream);
    // layer 2: final output to d_out
    run_layer(xbuf, (float*)d_out, src, dst, etype, rel_emb, W2, b2, Wr2, br2, attn2, g2, be2,
              hl, agg, ebuf, s_i, s_j, m_enc, denom, rs, u, c, N, E, R, stream);
}
